// FDHA_87797721464850
// MI455X (gfx1250) — matrix operand order repaired and checked by reading, not compiled, not run on hardware
//
#include <hip/hip_runtime.h>

#define PI2F 6.28318530717958647692f

typedef __attribute__((ext_vector_type(16))) _Float16 v16h;
typedef __attribute__((ext_vector_type(8)))  _Float16 v8h;
typedef __attribute__((ext_vector_type(4)))  _Float16 v4h;
typedef __attribute__((ext_vector_type(8)))  float    v8f;
typedef __attribute__((ext_vector_type(4)))  float    v4f;

static constexpr long NPIX = 65536;   // 256*256
static constexpr int  BB   = 4;

// ---------------------------------------------------------------------------
// WMMA fragment loaders (CDNA5 16x16x32 f16 layouts, from LDS)
// A 16x32: lane<16 -> M=lane, K 0..7 & 16..23 ; lane>=16 -> K 8..15 & 24..31
// B 32x16: lane<16 -> N=lane, K 0..7 & 16..23 ; lane>=16 -> K 8..15 & 24..31
// ---------------------------------------------------------------------------
__device__ __forceinline__ v16h ld_fragA(const _Float16* base, int ld) {
    int l = threadIdx.x & 31;
    const _Float16* p = base + (l & 15) * ld + ((l < 16) ? 0 : 8);
    v8h lo = *(const v8h*)p;
    v8h hi = *(const v8h*)(p + 16);
    v16h o;
#pragma unroll
    for (int i = 0; i < 8; ++i) { o[i] = lo[i]; o[i + 8] = hi[i]; }
    return o;
}
__device__ __forceinline__ v16h ld_fragB(const _Float16* base, int ld) {
    int l = threadIdx.x & 31;
    const _Float16* p = base + (l & 15) * ld + ((l < 16) ? 0 : 8);
    v8h lo = *(const v8h*)p;
    v8h hi = *(const v8h*)(p + 16);
    v16h o;
#pragma unroll
    for (int i = 0; i < 8; ++i) { o[i] = lo[i]; o[i + 8] = hi[i]; }
    return o;
}

// ---------------------------------------------------------------------------
// Generic WMMA GEMM:  C[M,N] = A[M,K] * B  (f32 in/out, f16 WMMA, f32 acc)
// A row-major (lda). transB=1: B stored [N,K] (ldb = row stride of N rows).
// transB=0: B stored [K,N]. Batch z: base += (z/subB)*outer + (z%subB)*inner.
// Block = 256 threads (8 waves), tile 64x64, each wave two 16x16 tiles.
// Fast path (all dims tile-multiples): float4 loads + packed v4h LDS stores.
// ---------------------------------------------------------------------------
__global__ void k_gemm(const float* __restrict__ A, const float* __restrict__ Bm,
                       float* __restrict__ C,
                       int M, int N, int K, int lda, int ldb, int ldc,
                       long aOut, long aIn, long bOut, long bIn, long cOut, long cIn,
                       int subB, int transB)
{
    __shared__ __align__(16) _Float16 sA[64 * 40];
    __shared__ __align__(16) _Float16 sB[64 * 40];
    int z = blockIdx.z;
    const float* Ab = A + (long)(z / subB) * aOut + (long)(z % subB) * aIn;
    const float* Bb = Bm + (long)(z / subB) * bOut + (long)(z % subB) * bIn;
    float*       Cb = C + (long)(z / subB) * cOut + (long)(z % subB) * cIn;
    int n0 = blockIdx.x * 64;
    int m0 = blockIdx.y * 64;
    int t = threadIdx.x;
    int wave = t >> 5, lane = t & 31;
    int tm = wave & 3, tn0 = (wave >> 2) << 1;
    bool full = ((M & 63) == 0) && ((N & 63) == 0) && ((K & 31) == 0);
    v8f acc0 = {}, acc1 = {};
    for (int k0 = 0; k0 < K; k0 += 32) {
        if (full) {
            // ---- unchecked vectorized staging ----
#pragma unroll
            for (int it = 0; it < 2; ++it) {
                int q = it * 256 + t;          // 512 float4 quads (64x32)
                int r = q >> 3;                // row 0..63
                int c4 = (q & 7) << 2;         // col {0,4,...,28}
                v4f va = *(const v4f*)(Ab + (long)(m0 + r) * lda + (k0 + c4));
                v4h ha;
#pragma unroll
                for (int u = 0; u < 4; ++u) ha[u] = (_Float16)va[u];
                *(v4h*)(sA + r * 40 + c4) = ha;
            }
            if (transB) {
#pragma unroll
                for (int it = 0; it < 2; ++it) {
                    int q = it * 256 + t;
                    int r = q >> 3;
                    int c4 = (q & 7) << 2;
                    v4f vb = *(const v4f*)(Bb + (long)(n0 + r) * ldb + (k0 + c4));
                    v4h hb;
#pragma unroll
                    for (int u = 0; u < 4; ++u) hb[u] = (_Float16)vb[u];
                    *(v4h*)(sB + r * 40 + c4) = hb;
                }
            } else {
#pragma unroll
                for (int it = 0; it < 2; ++it) {
                    int q = it * 256 + t;      // [k][n4] quads
                    int kk = q >> 4;           // 0..31
                    int n4 = (q & 15) << 2;    // 0..60
                    v4f vb = *(const v4f*)(Bb + (long)(k0 + kk) * ldb + (n0 + n4));
#pragma unroll
                    for (int u = 0; u < 4; ++u) sB[(n4 + u) * 40 + kk] = (_Float16)vb[u];
                }
            }
        } else {
            // ---- bounds-checked staging (padded tiles) ----
#pragma unroll
            for (int e0 = 0; e0 < 8; ++e0) {
                int e = e0 * 256 + t;
                int r = e >> 5, cc = e & 31;
                int gm = m0 + r, gn = n0 + r, gk = k0 + cc;
                float va = 0.f, vb = 0.f;
                if (gm < M && gk < K) va = Ab[(long)gm * lda + gk];
                if (gn < N && gk < K) {
                    if (transB) vb = Bb[(long)gn * ldb + gk];
                    else        vb = Bb[(long)gk * ldb + gn];
                }
                sA[r * 40 + cc] = (_Float16)va;
                sB[r * 40 + cc] = (_Float16)vb;
            }
        }
        if (k0 + 32 < K) {
            int pr = m0 + (t & 63);
            if (pr < M) __builtin_prefetch(Ab + (long)pr * lda + k0 + 32, 0, 1);
        }
        __syncthreads();
        v16h af  = ld_fragA(sA + tm * 16 * 40, 40);
        v16h bf0 = ld_fragB(sB + tn0 * 16 * 40, 40);
        v16h bf1 = ld_fragB(sB + (tn0 + 1) * 16 * 40, 40);
        acc0 = __builtin_amdgcn_wmma_f32_16x16x32_f16(false, af, false, bf0, (short)0, acc0, false, false);
        acc1 = __builtin_amdgcn_wmma_f32_16x16x32_f16(false, af, false, bf1, (short)0, acc1, false, false);
        __syncthreads();
    }
    int ln = lane & 15, mh = (lane >> 4) * 8;
#pragma unroll
    for (int r8 = 0; r8 < 8; ++r8) {
        int gm = m0 + tm * 16 + r8 + mh;
        int gn0 = n0 + tn0 * 16 + ln;
        if (gm < M) {
            if (gn0 < N)      Cb[(long)gm * ldc + gn0]      = acc0[r8];
            if (gn0 + 16 < N) Cb[(long)gm * ldc + gn0 + 16] = acc1[r8];
        }
    }
}

// ---------------------------------------------------------------------------
// 3x3 input conv (3 -> 64 channels, SAME)
// ---------------------------------------------------------------------------
__global__ void k_conv_in(const float* __restrict__ x, const float* __restrict__ w,
                          float* __restrict__ out)
{
    long i = (long)blockIdx.x * 256 + threadIdx.x;   // B*64*NPIX
    int px = i & 255, py = (int)((i >> 8) & 255);
    int oc = (int)((i >> 16) & 63);
    int b  = (int)(i >> 22);
    float s = 0.f;
    for (int ic = 0; ic < 3; ++ic) {
        const float* xp = x + ((long)(b * 3 + ic) << 16);
        const float* wp = w + (oc * 3 + ic) * 9;
#pragma unroll
        for (int dy = 0; dy < 3; ++dy) {
            int Y = py + dy - 1;
            if (Y < 0 || Y > 255) continue;
#pragma unroll
            for (int dx = 0; dx < 3; ++dx) {
                int X = px + dx - 1;
                if (X < 0 || X > 255) continue;
                s += wp[dy * 3 + dx] * xp[(Y << 8) + X];
            }
        }
    }
    out[i] = s;
}

// ---------------------------------------------------------------------------
// depthwise 3x3 (SAME), 16x16 output tile per block.
// Halo (18x18) is staged global->LDS with the gfx1250 async-DMA path
// (global_load_async_to_lds_b32, tracked by ASYNCcnt) - no VGPR round trip.
// ---------------------------------------------------------------------------
__global__ void k_dw3x3(const float* __restrict__ src, const float* __restrict__ wgt,
                        float* __restrict__ dst, int C)
{
    __shared__ __align__(16) float tile[18 * 20];
    int bc = blockIdx.z;                  // b*C + c
    int ty0 = blockIdx.y << 4, tx0 = blockIdx.x << 4;
    int t = threadIdx.x;
    int c = bc % C;
    // zero (covers out-of-image halo)
    for (int i = t; i < 18 * 20; i += 256) tile[i] = 0.f;
    __syncthreads();
    const float* sp = src + ((long)bc << 16);
    unsigned lbase = (unsigned)(unsigned long long)(uintptr_t)(&tile[0]);
    for (int e = t; e < 324; e += 256) {
        int ly = e / 18, lx = e - ly * 18;
        int gy = ty0 + ly - 1, gx = tx0 + lx - 1;
        if (gy >= 0 && gy < 256 && gx >= 0 && gx < 256) {
            unsigned loff = lbase + (unsigned)(ly * 20 + lx) * 4u;
            unsigned long long ga = (unsigned long long)(uintptr_t)(sp + (gy << 8) + gx);
            asm volatile("global_load_async_to_lds_b32 %0, %1, off"
                         :: "v"(loff), "v"(ga) : "memory");
        }
    }
    asm volatile("s_wait_asynccnt 0" ::: "memory");
    __syncthreads();
    const float* wp = wgt + c * 9;
    int ly = t >> 4, lx = t & 15;
    float s = 0.f;
#pragma unroll
    for (int dy = 0; dy < 3; ++dy)
#pragma unroll
        for (int dx = 0; dx < 3; ++dx)
            s += wp[dy * 3 + dx] * tile[(ly + dy) * 20 + (lx + dx)];
    dst[((long)bc << 16) + ((long)(ty0 + ly) << 8) + (tx0 + lx)] = s;
}

// ---------------------------------------------------------------------------
// 256-point radix-2 DIT FFT in LDS
// ---------------------------------------------------------------------------
__device__ void fft256(float* re, float* im, int t)
{
    for (int len = 2; len <= 256; len <<= 1) {
        int half = len >> 1;
        if (t < 128) {
            int grp = t / half, pos = t - grp * half;
            int i0 = grp * len + pos, i1 = i0 + half;
            float ang = -PI2F * (float)pos / (float)len;
            float sn, cs;
            __sincosf(ang, &sn, &cs);
            float xr = re[i1], xi = im[i1];
            float tr = xr * cs - xi * sn, ti = xr * sn + xi * cs;
            float ur = re[i0], ui = im[i0];
            re[i0] = ur + tr; im[i0] = ui + ti;
            re[i1] = ur - tr; im[i1] = ui - ti;
        }
        __syncthreads();
    }
}

// rows FFT: real input (b,c,h,:) -> complex
__global__ void k_fft_rows(const float* __restrict__ src, float2* __restrict__ dst,
                           int CH, int CSTR)
{
    __shared__ float re[256], im[256];
    int blk = blockIdx.x;                 // (b*CH + c)*256 + h
    int h = blk & 255, bc = blk >> 8;
    int b = bc / CH, c = bc % CH;
    const float* row = src + (((long)(b * CSTR + c) << 8) + h) * 256;
    int t = threadIdx.x;
    unsigned rv = __brev((unsigned)t) >> 24;
    re[rv] = row[t];
    im[rv] = 0.f;
    __syncthreads();
    fft256(re, im, t);
    dst[((long)bc * 256 + h) * 256 + t] = make_float2(re[t], im[t]);
}

// cols FFT + magnitude; mode 0: write float fa[bc][h][w]; mode 1: write
// packed (magbits<<32 | h*256+w) u64 keys for full-image argsort
__global__ void k_fft_cols(const float2* __restrict__ src, float* __restrict__ fa,
                           unsigned long long* __restrict__ pk, int mode)
{
    __shared__ float re[256], im[256];
    int blk = blockIdx.x;                 // bc*256 + w
    int w = blk & 255, bc = blk >> 8;
    const float2* col = src + ((long)bc << 16) + w;
    int t = threadIdx.x;
    unsigned rv = __brev((unsigned)t) >> 24;
    float2 v = col[(long)t * 256];
    re[rv] = v.x; im[rv] = v.y;
    __syncthreads();
    fft256(re, im, t);
    float mag = sqrtf(re[t] * re[t] + im[t] * im[t]);
    if (mode == 0) {
        fa[(((long)bc << 8) + t) * 256 + w] = mag;
    } else {
        unsigned kb = __float_as_uint(mag);
        pk[((long)bc << 16) + (t << 8) + w] =
            ((unsigned long long)kb << 32) | (unsigned)((t << 8) + w);
    }
}

// ---------------------------------------------------------------------------
// 256-element stable argsort (bitonic on packed u64 keys, LDS)
// ---------------------------------------------------------------------------
__device__ void bitonic256(unsigned long long* key, int t)
{
    for (int k = 2; k <= 256; k <<= 1) {
        for (int j = k >> 1; j > 0; j >>= 1) {
            int ixj = t ^ j;
            if (ixj > t) {
                unsigned long long a = key[t], b = key[ixj];
                bool asc = ((t & k) == 0);
                if (asc ? (a > b) : (a < b)) { key[t] = b; key[ixj] = a; }
            }
            __syncthreads();
        }
    }
}

__global__ void k_sort256_h(const float* __restrict__ fa, unsigned short* __restrict__ idxh)
{
    __shared__ unsigned long long key[256];
    int blk = blockIdx.x;                 // bc*256 + w
    int w = blk & 255, bc = blk >> 8;
    int t = threadIdx.x;
    float v = fa[(((long)bc << 8) + t) * 256 + w];
    key[t] = ((unsigned long long)__float_as_uint(v) << 32) | (unsigned)t;
    __syncthreads();
    bitonic256(key, t);
    idxh[(((long)bc << 8) + t) * 256 + w] = (unsigned short)(key[t] & 0xffffu);
}

__global__ void k_sort256_w(const float* __restrict__ fa, unsigned short* __restrict__ idxw)
{
    __shared__ unsigned long long key[256];
    int blk = blockIdx.x;                 // bc*256 + h
    int h = blk & 255, bc = blk >> 8;
    int t = threadIdx.x;
    float v = fa[(((long)bc << 8) + h) * 256 + t];
    key[t] = ((unsigned long long)__float_as_uint(v) << 32) | (unsigned)t;
    __syncthreads();
    bitonic256(key, t);
    idxw[(((long)bc << 8) + h) * 256 + t] = (unsigned short)(key[t] & 0xffffu);
}

// h-gather: ys[b,c,i,w] = y[b,c,idxh[b,c,i,w],w]   (c < 32)
__global__ void k_gather_h(const float* __restrict__ y, const unsigned short* __restrict__ idxh,
                           float* __restrict__ ys)
{
    long i = (long)blockIdx.x * 256 + threadIdx.x;   // B*32*NPIX
    int w = i & 255;
    long bc = i >> 16;
    int b = (int)(bc >> 5), c = (int)(bc & 31);
    int hs = idxh[i];
    ys[i] = y[((long)(b * 64 + c) << 16) + (hs << 8) + w];
}

// w-gather into y2 lower channels
__global__ void k_gather_w(const float* __restrict__ ys, const unsigned short* __restrict__ idxw,
                           float* __restrict__ y2)
{
    long i = (long)blockIdx.x * 256 + threadIdx.x;   // B*32*NPIX
    int hh = (int)((i >> 8) & 255);
    long bc = i >> 16;
    int b = (int)(bc >> 5), c = (int)(bc & 31);
    int ws = idxw[i];
    y2[((long)(b * 64 + c) << 16) + (hh << 8) + (i & 255)] =
        ys[(bc << 16) + (hh << 8) + ws];
}

// copy channels 32..63 (contiguous per batch)
__global__ void k_copy_upper(const float* __restrict__ src, float* __restrict__ dst)
{
    long i = (long)blockIdx.x * 256 + threadIdx.x;   // B*32*NPIX
    long b = i >> 21;
    long r = i & ((1L << 21) - 1);
    long a = ((b * 64 + 32) << 16) + r;
    dst[a] = src[a];
}

// ---------------------------------------------------------------------------
// 65536-element stable argsort: hybrid bitonic on u64 keys
// ---------------------------------------------------------------------------
__global__ void k_sort_local2048(unsigned long long* __restrict__ data)
{
    __shared__ unsigned long long s[2048];
    long base = (long)blockIdx.x * 2048;
    int t = threadIdx.x;
    for (int i = t; i < 2048; i += 256) s[i] = data[base + i];
    __syncthreads();
    for (int k = 2; k <= 2048; k <<= 1) {
        for (int j = k >> 1; j > 0; j >>= 1) {
            for (int e = t; e < 1024; e += 256) {
                int i = ((e & ~(j - 1)) << 1) | (e & (j - 1));
                int ixj = i | j;
                long gi = base + i;
                bool asc = ((gi & k) == 0);
                unsigned long long a = s[i], b = s[ixj];
                if (asc ? (a > b) : (a < b)) { s[i] = b; s[ixj] = a; }
            }
            __syncthreads();
        }
    }
    for (int i = t; i < 2048; i += 256) data[base + i] = s[i];
}

__global__ void k_sort_global_pass(unsigned long long* __restrict__ data, int K, int d)
{
    long e = (long)blockIdx.x * 256 + threadIdx.x;   // nprob * 32768 pairs
    long prob = e >> 15;
    long ee = e & 32767;
    long i = ((ee & ~(long)(d - 1)) << 1) | (ee & (d - 1));
    long ixj = i | d;
    long base = prob << 16;
    bool asc = (K == 65536) ? true : ((i & K) == 0);
    unsigned long long a = data[base + i], b = data[base + ixj];
    if (asc ? (a > b) : (a < b)) { data[base + i] = b; data[base + ixj] = a; }
}

__global__ void k_sort_local_finish(unsigned long long* __restrict__ data, int K)
{
    __shared__ unsigned long long s[2048];
    long base = (long)blockIdx.x * 2048;
    int t = threadIdx.x;
    for (int i = t; i < 2048; i += 256) s[i] = data[base + i];
    __syncthreads();
    long pin = base & 65535;
    bool asc = (K == 65536) ? true : ((pin & K) == 0);
    for (int j = 1024; j > 0; j >>= 1) {
        for (int e = t; e < 1024; e += 256) {
            int i = ((e & ~(j - 1)) << 1) | (e & (j - 1));
            int ixj = i | j;
            unsigned long long a = s[i], b = s[ixj];
            if (asc ? (a > b) : (a < b)) { s[i] = b; s[ixj] = a; }
        }
        __syncthreads();
    }
    for (int i = t; i < 2048; i += 256) data[base + i] = s[i];
}

__global__ void k_invperm(const unsigned long long* __restrict__ sortb, int* __restrict__ inv)
{
    long i = (long)blockIdx.x * 256 + threadIdx.x;   // B*64*NPIX
    long bc = i >> 16;
    unsigned p = (unsigned)(sortb[i] & 0xffffffffu);
    inv[(bc << 16) + p] = (int)(i & 65535);
}

__global__ void k_gather_perm(float* __restrict__ dst, const float* __restrict__ src,
                              const unsigned long long* __restrict__ sortb, long srcBStride)
{
    long i = (long)blockIdx.x * 256 + threadIdx.x;   // B*64*NPIX
    long bc = i >> 16;
    int b = (int)(bc >> 6), c = (int)(bc & 63);
    unsigned p = (unsigned)(sortb[i] & 0xffffffffu);
    dst[i] = src[(long)b * srcBStride + ((long)c << 16) + p];
}

// de-interleave (non-box heads): dst[b,h,r,n] = src[b, h*8+r/8, n*8 + r%8]
__global__ void k_pack(const float* __restrict__ src, float* __restrict__ dst)
{
    long i = (long)blockIdx.x * 256 + threadIdx.x;   // B*8*64*8192
    int n = i & 8191;
    int r = (int)((i >> 13) & 63);
    long bh = i >> 19;
    int b = (int)(bh >> 3), h = (int)(bh & 7);
    dst[i] = src[((long)(b * 64 + h * 8 + (r >> 3)) << 16) + ((long)n << 3) + (r & 7)];
}

__global__ void k_unpack(const float* __restrict__ srcPacked, float* __restrict__ dst)
{
    long i = (long)blockIdx.x * 256 + threadIdx.x;
    int n = i & 8191;
    int r = (int)((i >> 13) & 63);
    long bh = i >> 19;
    int b = (int)(bh >> 3), h = (int)(bh & 7);
    dst[((long)(b * 64 + h * 8 + (r >> 3)) << 16) + ((long)n << 3) + (r & 7)] = srcPacked[i];
}

// in-place row l2-normalization, rows at data + (row/nr)*bStride + (row%nr)*rowlen
__global__ void k_l2norm(float* __restrict__ data, int rowlen, int nr, long bStride)
{
    int row = blockIdx.x;
    int b = row / nr, r = row % nr;
    float* p = data + (long)b * bStride + (long)r * rowlen;
    __shared__ float red[256];
    __shared__ float sc_s;
    float s = 0.f;
    for (int i = threadIdx.x; i < rowlen; i += 256) { float v = p[i]; s += v * v; }
    red[threadIdx.x] = s;
    __syncthreads();
    for (int o = 128; o > 0; o >>= 1) {
        if (threadIdx.x < o) red[threadIdx.x] += red[threadIdx.x + o];
        __syncthreads();
    }
    if (threadIdx.x == 0) sc_s = 1.f / fmaxf(sqrtf(red[0]), 1e-12f);
    __syncthreads();
    float sc = sc_s;
    for (int i = threadIdx.x; i < rowlen; i += 256) p[i] *= sc;
}

// softmax_1 (no max-subtract, +1e-6 in denom) on 32 matrices of 64x64, pre-scaled by temp[h]
__global__ void k_softmax1(float* __restrict__ attn, const float* __restrict__ temp)
{
    int m = blockIdx.x;
    float tp = temp[m & 7];
    float* a = attn + (long)m * 4096;
    int r = threadIdx.x;
    if (r < 64) {
        float s = 0.f;
        for (int c = 0; c < 64; ++c) {
            float e = expf(a[r * 64 + c] * tp);
            a[r * 64 + c] = e;
            s += e;
        }
        float inv = 1.f / (s + 1e-6f);
        for (int c = 0; c < 64; ++c) a[r * 64 + c] *= inv;
    }
}

// proper softmax on 32 matrices of 8x8 (stored ldc=8, matrices spaced 64 floats)
__global__ void k_softmax_ca(float* __restrict__ attn, const float* __restrict__ temp)
{
    int m = blockIdx.x;
    float tp = temp[m & 7];
    float* a = attn + (long)m * 64;
    int r = threadIdx.x;
    if (r < 8) {
        float v[8], mx = -1e30f;
        for (int c = 0; c < 8; ++c) { v[c] = a[r * 8 + c] * tp; mx = fmaxf(mx, v[c]); }
        float s = 0.f;
        for (int c = 0; c < 8; ++c) { v[c] = expf(v[c] - mx); s += v[c]; }
        float inv = 1.f / s;
        for (int c = 0; c < 8; ++c) a[r * 8 + c] = v[c] * inv;
    }
}

// o12[bc,j] = o1[bc,inv[bc,j]] * o2[bc,inv[bc,j]]
__global__ void k_gather_mul(const float* __restrict__ o1, const float* __restrict__ o2,
                             const int* __restrict__ inv, float* __restrict__ o12)
{
    long i = (long)blockIdx.x * 256 + threadIdx.x;
    long base = (i >> 16) << 16;
    long q = base + inv[i];
    o12[i] = o1[q] * o2[q];
}

// scatter along w:  tmp[bc,h,idxw[bc,h,t]] = proj[(b*64+c),h,t]   (c<32)
__global__ void k_scatter_w(const float* __restrict__ proj, const unsigned short* __restrict__ idxw,
                            float* __restrict__ tmp)
{
    long i = (long)blockIdx.x * 256 + threadIdx.x;   // B*32*NPIX
    int t = i & 255, h = (int)((i >> 8) & 255);
    long bc = i >> 16;
    int b = (int)(bc >> 5), c = (int)(bc & 31);
    tmp[(bc << 16) + (h << 8) + idxw[i]] =
        proj[((long)(b * 64 + c) << 16) + (h << 8) + t];
}

// scatter along h:  outU[(b*64+c), idxh[bc,t,w], w] = tmp[bc,t,w]
__global__ void k_scatter_h(const float* __restrict__ tmp, const unsigned short* __restrict__ idxh,
                            float* __restrict__ outU)
{
    long i = (long)blockIdx.x * 256 + threadIdx.x;   // B*32*NPIX
    int w = i & 255;
    long bc = i >> 16;
    int b = (int)(bc >> 5), c = (int)(bc & 31);
    outU[((long)(b * 64 + c) << 16) + ((long)idxh[i] << 8) + w] = tmp[i];
}

// ca: out[b, h*8+cr, n] = sum_d attn[(b*8+h), cr, d] * v[b, 64+h*8+d, n]
__global__ void k_ca_av(const float* __restrict__ attn, const float* __restrict__ kv,
                        float* __restrict__ out)
{
    int row = blockIdx.x;                 // b*64 + ch
    int b = row >> 6, ch = row & 63;
    int h = ch >> 3, cr = ch & 7;
    float a[8];
#pragma unroll
    for (int d = 0; d < 8; ++d) a[d] = attn[(long)(b * 8 + h) * 64 + cr * 8 + d];
    const float* vb = kv + ((long)(b * 128 + 64 + h * 8) << 16);
    float* ob = out + ((long)row << 16);
    for (int n = threadIdx.x; n < 65536; n += 256) {
        float s = 0.f;
#pragma unroll
        for (int d = 0; d < 8; ++d) s += a[d] * vb[((long)d << 16) + n];
        ob[n] = s;
    }
}

// final: out = agg*para1[ch] + y*para2[ch]
__global__ void k_final(const float* __restrict__ agg, const float* __restrict__ y,
                        const float* __restrict__ p1, const float* __restrict__ p2,
                        float* __restrict__ out)
{
    long i = (long)blockIdx.x * 256 + threadIdx.x;   // B*64*NPIX
    int ch = (int)((i >> 16) & 63);
    out[i] = agg[i] * p1[ch] + y[i] * p2[ch];
}

// ---------------------------------------------------------------------------
// host orchestration
// ---------------------------------------------------------------------------
extern "C" void kernel_launch(void* const* d_in, const int* in_sizes, int n_in,
                              void* d_out, int out_size, void* d_ws, size_t ws_size,
                              hipStream_t stream)
{
    (void)in_sizes; (void)n_in; (void)out_size; (void)ws_size;
    const float* x        = (const float*)d_in[0];
    const float* y        = (const float*)d_in[1];
    const float* conv_in_w= (const float*)d_in[2];
    const float* qkv_w    = (const float*)d_in[3];
    const float* qkv_dw_w = (const float*)d_in[4];
    const float* proj_w   = (const float*)d_in[5];
    const float* temp     = (const float*)d_in[6];
    const float* para1    = (const float*)d_in[7];
    const float* para2    = (const float*)d_in[8];
    const float* ca_q_w   = (const float*)d_in[9];
    const float* ca_q_dw  = (const float*)d_in[10];
    const float* ca_kv_w  = (const float*)d_in[11];
    const float* ca_kv_dw = (const float*)d_in[12];
    const float* ca_proj_w= (const float*)d_in[13];
    const float* ca_temp  = (const float*)d_in[14];
    float* out = (float*)d_out;

    char* w = (char*)d_ws;
    const size_t S64  = (size_t)BB * 64 * NPIX * 4;        // 64 MB
    const size_t S32  = (size_t)BB * 32 * NPIX * 4;        // 32 MB
    size_t off = 0;
    auto alloc = [&](size_t b) { size_t o = off; off += (b + 255) & ~(size_t)255; return o; };
    size_t o_xconv = alloc(S64);
    size_t o_y2    = alloc(S64);
    size_t o_cplx  = alloc((size_t)BB * 64 * NPIX * 8);    // 128 MB complex
    size_t o_fa    = alloc(S32);                           // fa, later ys scratch
    size_t o_idxh  = alloc((size_t)BB * 32 * NPIX * 2);
    size_t o_idxw  = alloc((size_t)BB * 32 * NPIX * 2);
    size_t o_qkvA  = alloc((size_t)BB * 320 * NPIX * 4);   // 320 MB
    size_t o_qkvB  = alloc((size_t)BB * 320 * NPIX * 4);   // 320 MB
    size_t o_sort  = alloc((size_t)BB * 64 * NPIX * 8);    // 128 MB
    size_t o_inv   = alloc(S64);
    size_t o_g0 = alloc(S64), o_g1 = alloc(S64), o_g2 = alloc(S64);
    size_t o_g3 = alloc(S64), o_g4 = alloc(S64);
    size_t o_attn1 = alloc(32 * 4096 * 4);
    size_t o_attn2 = alloc(32 * 4096 * 4);
    size_t o_cattn = alloc(32 * 64 * 4);

    float* xconv = (float*)(w + o_xconv);
    float* y2    = (float*)(w + o_y2);
    float2* cplx = (float2*)(w + o_cplx);
    float* fa    = (float*)(w + o_fa);
    unsigned short* idxh = (unsigned short*)(w + o_idxh);
    unsigned short* idxw = (unsigned short*)(w + o_idxw);
    float* qkvA  = (float*)(w + o_qkvA);
    float* qkvB  = (float*)(w + o_qkvB);
    unsigned long long* sortb = (unsigned long long*)(w + o_sort);
    int*   inv   = (int*)(w + o_inv);
    float* g0 = (float*)(w + o_g0), *g1 = (float*)(w + o_g1), *g2 = (float*)(w + o_g2);
    float* g3 = (float*)(w + o_g3), *g4 = (float*)(w + o_g4);
    float* attn1 = (float*)(w + o_attn1);
    float* attn2 = (float*)(w + o_attn2);
    float* cattn = (float*)(w + o_cattn);
    // aliases (regions provably dead at time of reuse)
    float* p1 = qkvA;
    float* p2 = (float*)((char*)qkvA + S64);
    float* p3 = (float*)((char*)qkvA + 2 * S64);
    float* o1      = qkvB;
    float* o2p     = (float*)((char*)qkvB + S64);
    float* o2      = (float*)((char*)qkvB + 2 * S64);
    float* o12     = (float*)((char*)qkvB + 3 * S64);
    float* projout = (float*)((char*)qkvB + 4 * S64);
    float* tmpw    = g0;
    float* outU    = g1;
    float* ca_qA   = g2;
    float* ca_q    = g3;
    float* ca_kvA  = qkvA;                    // 128 MB
    float* ca_kv   = (float*)((char*)qkvA + 2 * S64);
    float* ca_out  = g4;
    float* agg     = (float*)(w + o_sort);

    auto gemm = [&](const float* A, const float* Bm, float* C, int M, int N, int K,
                    int lda, int ldb, int ldc,
                    long aOut, long aIn, long bOut, long bIn, long cOut, long cIn,
                    int subB, int transB, int nb) {
        dim3 g((N + 63) / 64, (M + 63) / 64, nb);
        k_gemm<<<g, 256, 0, stream>>>(A, Bm, C, M, N, K, lda, ldb, ldc,
                                      aOut, aIn, bOut, bIn, cOut, cIn, subB, transB);
    };
    const long NP = NPIX;

    // 1) input conv
    k_conv_in<<<dim3(BB * 64 * NPIX / 256), 256, 0, stream>>>(x, conv_in_w, xconv);

    // 2) |fft2(y_half)| and row/col argsorts
    k_fft_rows<<<dim3(BB * 32 * 256), 256, 0, stream>>>(y, cplx, 32, 64);
    k_fft_cols<<<dim3(BB * 32 * 256), 256, 0, stream>>>(cplx, fa, nullptr, 0);
    k_sort256_h<<<dim3(BB * 32 * 256), 256, 0, stream>>>(fa, idxh);
    k_sort256_w<<<dim3(BB * 32 * 256), 256, 0, stream>>>(fa, idxw);
    // gather (fa region reused as ys scratch)
    k_gather_h<<<dim3(BB * 32 * NPIX / 256), 256, 0, stream>>>(y, idxh, fa);
    k_gather_w<<<dim3(BB * 32 * NPIX / 256), 256, 0, stream>>>(fa, idxw, y2);
    k_copy_upper<<<dim3(BB * 32 * NPIX / 256), 256, 0, stream>>>(y, y2);

    // 3) qkv = dw3x3(1x1(y2))
    gemm(qkv_w, y2, qkvA, 320, 65536, 64, 64, 65536, 65536,
         0, 0, 64 * NP, 0, 320 * NP, 0, 1, 0, BB);
    k_dw3x3<<<dim3(16, 16, BB * 320), 256, 0, stream>>>(qkvA, qkv_dw_w, qkvB, 320);

    // 4) |fft2(v)| -> packed keys -> full 65536 argsort
    k_fft_rows<<<dim3(BB * 64 * 256), 256, 0, stream>>>(qkvB + 256 * NP, cplx, 64, 320);
    k_fft_cols<<<dim3(BB * 64 * 256), 256, 0, stream>>>(cplx, nullptr, sortb, 1);
    k_sort_local2048<<<dim3(BB * 64 * NPIX / 2048), 256, 0, stream>>>(sortb);
    for (int K = 4096; K <= 65536; K <<= 1) {
        for (int d = K >> 1; d >= 2048; d >>= 1)
            k_sort_global_pass<<<dim3(BB * 64 * 32768 / 256), 256, 0, stream>>>(sortb, K, d);
        k_sort_local_finish<<<dim3(BB * 64 * NPIX / 2048), 256, 0, stream>>>(sortb, K);
    }
    k_invperm<<<dim3(BB * 64 * NPIX / 256), 256, 0, stream>>>(sortb, inv);

    // 5) gather q1,k1,q2,k2,v through the permutation
    float* gs[5] = {g0, g1, g2, g3, g4};
    for (int t = 0; t < 5; ++t)
        k_gather_perm<<<dim3(BB * 64 * NPIX / 256), 256, 0, stream>>>(
            gs[t], qkvB + (long)t * 64 * NP, sortb, 320 * NP);

    // 6) de-interleave for the non-box attention
    k_pack<<<dim3(BB * 64 * NPIX / 256), 256, 0, stream>>>(g2, p1);
    k_pack<<<dim3(BB * 64 * NPIX / 256), 256, 0, stream>>>(g3, p2);
    k_pack<<<dim3(BB * 64 * NPIX / 256), 256, 0, stream>>>(g4, p3);

    // 7) l2norm rows (8192-long, 2048 rows each)
    k_l2norm<<<dim3(2048), 256, 0, stream>>>(g0, 8192, 2048, 0);
    k_l2norm<<<dim3(2048), 256, 0, stream>>>(g1, 8192, 2048, 0);
    k_l2norm<<<dim3(2048), 256, 0, stream>>>(p1, 8192, 2048, 0);
    k_l2norm<<<dim3(2048), 256, 0, stream>>>(p2, 8192, 2048, 0);

    // 8) attn = q . k^T (WMMA), softmax_1, out = attn . v (WMMA)
    gemm(g0, g1, attn1, 64, 64, 8192, 8192, 8192, 64,
         64 * 8192L, 0, 64 * 8192L, 0, 4096, 0, 1, 1, 32);
    gemm(p1, p2, attn2, 64, 64, 8192, 8192, 8192, 64,
         64 * 8192L, 0, 64 * 8192L, 0, 4096, 0, 1, 1, 32);
    k_softmax1<<<dim3(32), 256, 0, stream>>>(attn1, temp);
    k_softmax1<<<dim3(32), 256, 0, stream>>>(attn2, temp);
    gemm(attn1, g4, o1, 64, 8192, 64, 64, 8192, 8192,
         4096, 0, 64 * 8192L, 0, 64 * 8192L, 0, 1, 0, 32);
    gemm(attn2, p3, o2p, 64, 8192, 64, 64, 8192, 8192,
         4096, 0, 64 * 8192L, 0, 64 * 8192L, 0, 1, 0, 32);
    k_unpack<<<dim3(BB * 64 * NPIX / 256), 256, 0, stream>>>(o2p, o2);

    // 9) un-permute + multiply, projection
    k_gather_mul<<<dim3(BB * 64 * NPIX / 256), 256, 0, stream>>>(o1, o2, inv, o12);
    gemm(proj_w, o12, projout, 64, 65536, 64, 64, 65536, 65536,
         0, 0, 64 * NP, 0, 64 * NP, 0, 1, 0, BB);

    // 10) unsort lower half channels
    k_scatter_w<<<dim3(BB * 32 * NPIX / 256), 256, 0, stream>>>(projout, idxw, tmpw);
    k_scatter_h<<<dim3(BB * 32 * NPIX / 256), 256, 0, stream>>>(tmpw, idxh, outU);
    k_copy_upper<<<dim3(BB * 32 * NPIX / 256), 256, 0, stream>>>(projout, outU);

    // 11) channel cross attention
    gemm(ca_q_w, outU, ca_qA, 64, 65536, 64, 64, 65536, 65536,
         0, 0, 64 * NP, 0, 64 * NP, 0, 1, 0, BB);
    k_dw3x3<<<dim3(16, 16, BB * 64), 256, 0, stream>>>(ca_qA, ca_q_dw, ca_q, 64);
    gemm(ca_kv_w, xconv, ca_kvA, 128, 65536, 64, 64, 65536, 65536,
         0, 0, 64 * NP, 0, 128 * NP, 0, 1, 0, BB);
    k_dw3x3<<<dim3(16, 16, BB * 128), 256, 0, stream>>>(ca_kvA, ca_kv_dw, ca_kv, 128);
    k_l2norm<<<dim3(256), 256, 0, stream>>>(ca_q, 65536, 256, 0);
    k_l2norm<<<dim3(256), 256, 0, stream>>>(ca_kv, 65536, 64, 128 * NP);
    // 8x8 attention over K=65536 (padded 16x16 WMMA tile, bounds-checked path)
    gemm(ca_q, ca_kv, cattn, 8, 8, 65536, 65536, 65536, 8,
         64 * NP, 8 * NP, 128 * NP, 8 * NP, 512, 64, 8, 1, 32);
    k_softmax_ca<<<dim3(32), 256, 0, stream>>>(cattn, ca_temp);
    k_ca_av<<<dim3(BB * 64), 256, 0, stream>>>(cattn, ca_kv, ca_out);
    gemm(ca_proj_w, ca_out, agg, 64, 65536, 64, 64, 65536, 65536,
         0, 0, 64 * NP, 0, 64 * NP, 0, 1, 0, BB);

    // 12) final blend
    k_final<<<dim3(BB * 64 * NPIX / 256), 256, 0, stream>>>(agg, y, para1, para2, out);
}
